// CausalSelfAttention_50019189129241
// MI455X (gfx1250) — matrix operand order repaired and checked by reading, not compiled, not run on hardware
//
#include <hip/hip_runtime.h>

// ---------------------------------------------------------------------------
// CDNA5 (gfx1250, wave32) causal self-attention:
//   qkv = x @ Wqkv^T ; flash-attention (causal, online softmax) ; y @ Wproj^T
// All matmuls on v_wmma_f32_16x16x32_bf16 (bf16 inputs, fp32 accum).
// ---------------------------------------------------------------------------

typedef unsigned short u16;
typedef __attribute__((ext_vector_type(16))) __bf16       v16bf;
typedef __attribute__((ext_vector_type(8)))  float        v8f;
typedef __attribute__((ext_vector_type(4)))  unsigned int v4u;

union FragBF {            // 16 bf16 per lane = 8 VGPRs (A or B fragment)
    v16bf bf;
    v4u   u[2];
};

static __device__ inline v8f wmma_bf16(const FragBF& a, const FragBF& b, v8f c) {
    // (neg_a, A, neg_b, B, c_mod, C, reuse_a, reuse_b)
    return __builtin_amdgcn_wmma_f32_16x16x32_bf16(false, a.bf, false, b.bf,
                                                   (short)0, c, false, false);
}

static __device__ inline u16 f2bf(float f) {   // round-to-nearest-even f32->bf16
    unsigned u = __float_as_uint(f);
    u += 0x7FFFu + ((u >> 16) & 1u);
    return (u16)(u >> 16);
}

#define T_SEQ 2048
#define C_DIM 1024
#define H_NUM 16
#define D_HEAD 64
#define BT_ROWS 8192        // B * T

// ---------------------------------------------------------------------------
// fp32 -> bf16 conversion (grid-stride)
// ---------------------------------------------------------------------------
__global__ void cvt_f32_bf16(const float* __restrict__ src, u16* __restrict__ dst, int n) {
    for (int i = blockIdx.x * blockDim.x + threadIdx.x; i < n; i += gridDim.x * blockDim.x)
        dst[i] = f2bf(src[i]);
}

// ---------------------------------------------------------------------------
// QKV GEMM: A[8192,1024] (bf16) x Wqkv[3072,1024]^T -> scatter into
//   Q[b,h,t,d], K[b,h,t,d], Vt[b,h,d,t]   (all bf16)
// Block: 256 thr = 8 waves, block tile 128(M) x 128(N); wave tile 32 x 64.
// ---------------------------------------------------------------------------
__global__ __launch_bounds__(256)
void gemm_qkv(const u16* __restrict__ A, const u16* __restrict__ W,
              u16* __restrict__ Qb, u16* __restrict__ Kb, u16* __restrict__ Vtb) {
    const int K = C_DIM;
    const int lane = threadIdx.x & 31, wave = threadIdx.x >> 5;
    const int half = lane >> 4, r = lane & 15;
    const int wm = wave >> 1, wn = wave & 1;
    const int m0 = blockIdx.y * 128 + wm * 32;
    const int n0 = blockIdx.x * 128 + wn * 64;

    v8f acc[2][4];
#pragma unroll
    for (int i = 0; i < 2; ++i)
#pragma unroll
        for (int j = 0; j < 4; ++j)
#pragma unroll
            for (int e = 0; e < 8; ++e) acc[i][j][e] = 0.f;

    for (int k0 = 0; k0 < K; k0 += 32) {
        FragBF a[2], b[4];
#pragma unroll
        for (int i = 0; i < 2; ++i) {
            const u16* ap = A + (size_t)(m0 + i * 16 + r) * K + k0 + half * 8;
            a[i].u[0] = *(const v4u*)ap;          // K = k0+half*8   .. +7
            a[i].u[1] = *(const v4u*)(ap + 16);   // K = k0+16+half*8.. +7
            if (k0 + 32 < K) __builtin_prefetch(ap + 32, 0, 0);
        }
#pragma unroll
        for (int j = 0; j < 4; ++j) {
            const u16* bp = W + (size_t)(n0 + j * 16 + r) * K + k0 + half * 8;
            b[j].u[0] = *(const v4u*)bp;          // K = k0+half*8 .. +7, k0+16+half*8 .. +7
            b[j].u[1] = *(const v4u*)(bp + 16);
        }
#pragma unroll
        for (int i = 0; i < 2; ++i)
#pragma unroll
            for (int j = 0; j < 4; ++j)
                acc[i][j] = wmma_bf16(a[i], b[j], acc[i][j]);
    }

    // epilogue: D layout row = half*8+e, col = r within each 16x16 tile
#pragma unroll
    for (int i = 0; i < 2; ++i)
#pragma unroll
        for (int j = 0; j < 4; ++j)
#pragma unroll
            for (int e = 0; e < 8; ++e) {
                int row = m0 + i * 16 + half * 8 + e;    // bt index
                int col = n0 + j * 16 + r;               // o in [0,3072)
                int bb  = row >> 11;                     // T = 2048
                int tt  = row & (T_SEQ - 1);
                int sec = col >> 10;                     // 0=q 1=k 2=v
                int oo  = col & (C_DIM - 1);
                int hh  = oo >> 6;
                int dd  = oo & (D_HEAD - 1);
                int bh  = (bb << 4) + hh;
                u16 v = f2bf(acc[i][j][e]);
                if (sec == 0)      Qb [((size_t)bh * T_SEQ + tt) * D_HEAD + dd] = v;
                else if (sec == 1) Kb [((size_t)bh * T_SEQ + tt) * D_HEAD + dd] = v;
                else               Vtb[((size_t)bh * D_HEAD + dd) * T_SEQ + tt] = v;
            }
}

// ---------------------------------------------------------------------------
// Flash attention (causal). Block = 128 thr = 4 waves; each wave: 16 q rows.
// grid = (B*H, T/64). Per 32-key chunk: 4 WMMA (QK^T, Kdim=64) + online
// softmax + P relayout via wave-private LDS + 4 WMMA (P*V, Kdim=32 keys).
// ---------------------------------------------------------------------------
__global__ __launch_bounds__(128)
void flash_attn(const u16* __restrict__ Qb, const u16* __restrict__ Kb,
                const u16* __restrict__ Vtb, u16* __restrict__ Yb) {
    __shared__ __align__(16) u16 plds[4][16 * 32];   // per-wave P tile (16x32 bf16)

    const int bh   = blockIdx.x;              // b*16 + h
    const int b    = bh >> 4;
    const int h    = bh & 15;
    const int lane = threadIdx.x & 31, wave = threadIdx.x >> 5;
    const int half = lane >> 4, r = lane & 15;
    const int q0   = blockIdx.y * 64 + wave * 16;
    const float scale = 0.125f;               // 1/sqrt(64)

    // Q fragments (A layout), Kdim chunks d=0..31 and d=32..63
    FragBF qf0, qf1;
    {
        const u16* qp = Qb + ((size_t)bh * T_SEQ + q0 + r) * D_HEAD;
        qf0.u[0] = *(const v4u*)(qp + half * 8);
        qf0.u[1] = *(const v4u*)(qp + 16 + half * 8);
        qf1.u[0] = *(const v4u*)(qp + 32 + half * 8);
        qf1.u[1] = *(const v4u*)(qp + 48 + half * 8);
    }

    v8f o[4];
    float mrun[8], lrun[8];
#pragma unroll
    for (int j = 0; j < 4; ++j)
#pragma unroll
        for (int e = 0; e < 8; ++e) o[j][e] = 0.f;
#pragma unroll
    for (int e = 0; e < 8; ++e) { mrun[e] = -1e30f; lrun[e] = 0.f; }

    const int qhi = q0 + 15;
    for (int k0 = 0; k0 <= qhi; k0 += 32) {
        // ---- S = Q K^T for two 16-key subtiles -------------------------
        v8f s[2];
#pragma unroll
        for (int j = 0; j < 2; ++j) {
            FragBF kf0, kf1;
            const u16* kp = Kb + ((size_t)bh * T_SEQ + k0 + j * 16 + r) * D_HEAD + half * 8;
            kf0.u[0] = *(const v4u*)kp;        kf0.u[1] = *(const v4u*)(kp + 16);
            kf1.u[0] = *(const v4u*)(kp + 32); kf1.u[1] = *(const v4u*)(kp + 48);
            v8f z;
#pragma unroll
            for (int e = 0; e < 8; ++e) z[e] = 0.f;
            z = wmma_bf16(qf0, kf0, z);
            z = wmma_bf16(qf1, kf1, z);
            s[j] = z;
        }

        // ---- causal mask + scale --------------------------------------
#pragma unroll
        for (int j = 0; j < 2; ++j)
#pragma unroll
            for (int e = 0; e < 8; ++e) {
                int kg = k0 + j * 16 + r;         // key (column = lane&15)
                int qg = q0 + half * 8 + e;       // query (row)
                float v = s[j][e] * scale;
                s[j][e] = (kg <= qg) ? v : -1e30f;
            }

        // ---- online softmax -------------------------------------------
        float cmax[8], corr[8];
#pragma unroll
        for (int e = 0; e < 8; ++e) {
            float m = fmaxf(s[0][e], s[1][e]);
#pragma unroll
            for (int off = 1; off < 16; off <<= 1)
                m = fmaxf(m, __shfl_xor(m, off, 16));
            float mnew = fmaxf(mrun[e], m);
            corr[e] = __expf(mrun[e] - mnew);
            mrun[e] = mnew;
        }
#pragma unroll
        for (int j = 0; j < 2; ++j)
#pragma unroll
            for (int e = 0; e < 8; ++e)
                s[j][e] = __expf(s[j][e] - mrun[e]);
#pragma unroll
        for (int e = 0; e < 8; ++e) {
            float rs = s[0][e] + s[1][e];
#pragma unroll
            for (int off = 1; off < 16; off <<= 1)
                rs += __shfl_xor(rs, off, 16);
            lrun[e] = lrun[e] * corr[e] + rs;
        }
#pragma unroll
        for (int j = 0; j < 4; ++j)
#pragma unroll
            for (int e = 0; e < 8; ++e) o[j][e] *= corr[e];

        // ---- P: D-layout -> A-fragment via wave-private LDS -----------
        u16* myp = &plds[wave][0];
#pragma unroll
        for (int j = 0; j < 2; ++j)
#pragma unroll
            for (int e = 0; e < 8; ++e)
                myp[(half * 8 + e) * 32 + j * 16 + r] = f2bf(s[j][e]);
        asm volatile("s_wait_dscnt 0x0" ::: "memory");
        FragBF pa;
        {
            const u16* pp = myp + r * 32 + half * 8;
            pa.u[0] = *(const v4u*)pp;
            pa.u[1] = *(const v4u*)(pp + 16);
        }
        asm volatile("s_wait_dscnt 0x0" ::: "memory");

        // ---- O += P * V  (V^T rows are K-major => contiguous B frags) -
#pragma unroll
        for (int j2 = 0; j2 < 4; ++j2) {
            const u16* vp = Vtb + ((size_t)bh * D_HEAD + j2 * 16 + r) * T_SEQ + k0 + half * 8;
            FragBF vf;
            vf.u[0] = *(const v4u*)vp;
            vf.u[1] = *(const v4u*)(vp + 16);
            o[j2] = wmma_bf16(pa, vf, o[j2]);
        }
    }

    // ---- normalize and store y[b,t,c] (bf16) ---------------------------
#pragma unroll
    for (int e = 0; e < 8; ++e) lrun[e] = 1.f / lrun[e];
#pragma unroll
    for (int j2 = 0; j2 < 4; ++j2)
#pragma unroll
        for (int e = 0; e < 8; ++e) {
            int t = q0 + half * 8 + e;
            int c = h * D_HEAD + j2 * 16 + r;
            Yb[((size_t)b * T_SEQ + t) * C_DIM + c] = f2bf(o[j2][e] * lrun[e]);
        }
}

// ---------------------------------------------------------------------------
// Output projection: Y[8192,1024] (bf16) x Wproj[1024,1024]^T -> fp32 out
// ---------------------------------------------------------------------------
__global__ __launch_bounds__(256)
void gemm_proj(const u16* __restrict__ A, const u16* __restrict__ W,
               float* __restrict__ Out) {
    const int K = C_DIM, N = C_DIM;
    const int lane = threadIdx.x & 31, wave = threadIdx.x >> 5;
    const int half = lane >> 4, r = lane & 15;
    const int wm = wave >> 1, wn = wave & 1;
    const int m0 = blockIdx.y * 128 + wm * 32;
    const int n0 = blockIdx.x * 128 + wn * 64;

    v8f acc[2][4];
#pragma unroll
    for (int i = 0; i < 2; ++i)
#pragma unroll
        for (int j = 0; j < 4; ++j)
#pragma unroll
            for (int e = 0; e < 8; ++e) acc[i][j][e] = 0.f;

    for (int k0 = 0; k0 < K; k0 += 32) {
        FragBF a[2], b[4];
#pragma unroll
        for (int i = 0; i < 2; ++i) {
            const u16* ap = A + (size_t)(m0 + i * 16 + r) * K + k0 + half * 8;
            a[i].u[0] = *(const v4u*)ap;
            a[i].u[1] = *(const v4u*)(ap + 16);
            if (k0 + 32 < K) __builtin_prefetch(ap + 32, 0, 0);
        }
#pragma unroll
        for (int j = 0; j < 4; ++j) {
            const u16* bp = W + (size_t)(n0 + j * 16 + r) * K + k0 + half * 8;
            b[j].u[0] = *(const v4u*)bp;
            b[j].u[1] = *(const v4u*)(bp + 16);
        }
#pragma unroll
        for (int i = 0; i < 2; ++i)
#pragma unroll
            for (int j = 0; j < 4; ++j)
                acc[i][j] = wmma_bf16(a[i], b[j], acc[i][j]);
    }

#pragma unroll
    for (int i = 0; i < 2; ++i)
#pragma unroll
        for (int j = 0; j < 4; ++j)
#pragma unroll
            for (int e = 0; e < 8; ++e) {
                int row = m0 + i * 16 + half * 8 + e;
                int col = n0 + j * 16 + r;
                Out[(size_t)row * N + col] = acc[i][j][e];
            }
}

// ---------------------------------------------------------------------------
extern "C" void kernel_launch(void* const* d_in, const int* in_sizes, int n_in,
                              void* d_out, int out_size, void* d_ws, size_t ws_size,
                              hipStream_t stream) {
    (void)in_sizes; (void)n_in; (void)out_size; (void)ws_size;
    const float* x     = (const float*)d_in[0];   // [4,2048,1024]
    const float* wqkv  = (const float*)d_in[1];   // [3072,1024]
    const float* wproj = (const float*)d_in[2];   // [1024,1024]
    float*       out   = (float*)d_out;           // [4,2048,1024]

    // workspace layout (bf16 u16 elements), ~88 MiB total (L2-resident)
    u16* xb     = (u16*)d_ws;
    u16* wqkvb  = xb    + (size_t)BT_ROWS * C_DIM;       // 8,388,608
    u16* wprojb = wqkvb + (size_t)3 * C_DIM * C_DIM;     // 3,145,728
    u16* qb     = wprojb + (size_t)C_DIM * C_DIM;        // 1,048,576
    u16* kb     = qb    + (size_t)BT_ROWS * C_DIM;
    u16* vtb    = kb    + (size_t)BT_ROWS * C_DIM;
    u16* yb     = vtb   + (size_t)BT_ROWS * C_DIM;

    cvt_f32_bf16<<<4096, 256, 0, stream>>>(x,     xb,     BT_ROWS * C_DIM);
    cvt_f32_bf16<<<4096, 256, 0, stream>>>(wqkv,  wqkvb,  3 * C_DIM * C_DIM);
    cvt_f32_bf16<<<4096, 256, 0, stream>>>(wproj, wprojb, C_DIM * C_DIM);

    gemm_qkv<<<dim3(3 * C_DIM / 128, BT_ROWS / 128), 256, 0, stream>>>(
        xb, wqkvb, qb, kb, vtb);

    flash_attn<<<dim3(4 * H_NUM, T_SEQ / 64), 128, 0, stream>>>(qb, kb, vtb, yb);

    gemm_proj<<<dim3(C_DIM / 128, BT_ROWS / 128), 256, 0, stream>>>(
        yb, wprojb, out);
}
